// S5R_7516192768286
// MI455X (gfx1250) — matrix operand order repaired and checked by reading, not compiled, not run on hardware
//
#include <hip/hip_runtime.h>
#include <math.h>

// Problem constants (match reference)
#define Bc 8
#define Hc 512
#define Pc 512
#define Lc 2048

// GEMM tiling
#define BM 128
#define BN 128
#define BKT 64
#define TP  72      // padded LDS stride (halves): 144B, 16B-aligned, bank-friendly
#define NTHREADS 256

typedef __attribute__((ext_vector_type(16))) __bf16 v16bf;
typedef __attribute__((ext_vector_type(8)))  float  v8f;
typedef __attribute__((ext_vector_type(4)))  unsigned int v4u;
typedef __attribute__((ext_vector_type(4)))  unsigned int u32x4;
typedef __attribute__((ext_vector_type(8)))  int i32x8;
typedef __attribute__((ext_vector_type(4)))  int i32x4;

union Frag { v4u q[2]; v16bf v; };

#ifndef __has_builtin
#define __has_builtin(x) 0
#endif
#if __has_builtin(__builtin_amdgcn_tensor_load_to_lds) && \
    __has_builtin(__builtin_amdgcn_s_wait_tensorcnt)
#define USE_TDM 1
#if defined(__clang_major__) && (__clang_major__ >= 23)
#define TDM_CALL(g0, g1, g2, g3) \
    __builtin_amdgcn_tensor_load_to_lds((g0), (g1), (g2), (g3), (i32x8){0,0,0,0,0,0,0,0}, 0)
#else
#define TDM_CALL(g0, g1, g2, g3) \
    __builtin_amdgcn_tensor_load_to_lds((g0), (g1), (g2), (g3), 0)
#endif
#endif

__device__ __forceinline__ unsigned short f2bf(float f) {
    union { float f; unsigned int u; } x; x.f = f;
    unsigned int r = x.u + 0x7FFFu + ((x.u >> 16) & 1u);   // RNE
    return (unsigned short)(r >> 16);
}

__device__ __forceinline__ float gelu_exact(float x) {
    return 0.5f * x * (1.0f + erff(x * 0.70710678118654752f));
}

#ifdef USE_TDM
// Issue a TDM copy of a BM x BKT bf16 tile from packed panel W (Mtot x Kdim,
// row-major, K contiguous) into LDS at ldsOff, padded to TP-half row stride.
// D# built per CDNA5 ISA ch.8: 2-D tile, data_size=2B,
// pad_enable, pad_interval=4 (every 32 DWORDs = one 128B row), pad_amount=3 (4 DWORDs = 16B).
__device__ __forceinline__ void tdm_load_a_tile(const unsigned short* W,
                                                unsigned ldsOff,
                                                int m0, int kt,
                                                int Kdim, int Mtot)
{
    unsigned long long gaddr = (unsigned long long)(uintptr_t)W +
        ((unsigned long long)(unsigned)m0 * (unsigned)Kdim + (unsigned)kt) * 2ull;
    u32x4 g0 = {
        1u,                                             // count=1, is_restore=0
        ldsOff,                                         // lds_addr
        (unsigned)(gaddr & 0xFFFFFFFFull),              // global_addr[31:0]
        ((unsigned)((gaddr >> 32) & 0x1FFFFFFull)) | 0x80000000u  // [56:32] | type=2<<30
    };
    i32x8 g1 = {
        (int)((1u << 16) | (1u << 20) | (4u << 22) | (3u << 25)), // data_size=1(2B), pad_en, intv=4, amt=3
        (int)(((unsigned)Kdim & 0xFFFFu) << 16),                  // tensor_dim0 lo16 @ bits[63:48]
        (int)((((unsigned)Kdim >> 16) & 0xFFFFu) |
              (((unsigned)Mtot & 0xFFFFu) << 16)),                // dim0 hi16 | tensor_dim1 lo16
        (int)((((unsigned)Mtot >> 16) & 0xFFFFu) |
              ((unsigned)BKT << 16)),                             // dim1 hi16 | tile_dim0=64
        (int)BM,                                                  // tile_dim1=128 (tile_dim2=0)
        (int)Kdim,                                                // tensor_dim0_stride lo32
        0, 0                                                      // stride hi, dim1_stride (unused 2D)
    };
    i32x4 gz = {0, 0, 0, 0};
    TDM_CALL(g0, g1, gz, gz);
}
#endif

// ---------------------------------------------------------------------------
// Pack kernels: weights -> bf16 panels (once; removes cvt from GEMM hot loops)
//   W_bu  (2P x H):  rows [0,P) = Re(B_bar), rows [P,2P) = Im(B_bar)
//   W_out (H x 2P):  cols [0,P) = Re(C),     cols [P,2P) = -Im(C)
// ---------------------------------------------------------------------------
__global__ __launch_bounds__(NTHREADS)
void pack_wbu_kernel(const float* __restrict__ Bbar, unsigned short* __restrict__ W)
{
    int idx = blockIdx.x * NTHREADS + threadIdx.x;     // 2P*H total
    int m = idx >> 9;                                  // / Hc
    int k = idx & (Hc - 1);
    int comp = m >> 9;                                 // m >= Pc
    int p = m & (Pc - 1);
    W[idx] = f2bf(Bbar[(((size_t)p) * Hc + (size_t)k) * 2 + comp]);
}

__global__ __launch_bounds__(NTHREADS)
void pack_wout_kernel(const float* __restrict__ Cri, unsigned short* __restrict__ W)
{
    int idx = blockIdx.x * NTHREADS + threadIdx.x;     // H*2P total
    int m = idx >> 10;                                 // / (2*Pc)
    int k = idx & (2 * Pc - 1);
    int comp = k >> 9;
    int p = k & (Pc - 1);
    float v = Cri[(((size_t)m) * Pc + (size_t)p) * 2 + comp];
    W[idx] = f2bf(comp ? -v : v);
}

// ---------------------------------------------------------------------------
// Kernel 1: Bu = B_bar @ u  (complex weights x real activations)
// M axis covers 2P rows (Re then Im). A tiles staged by TDM from W_bu panel.
// ---------------------------------------------------------------------------
__global__ __launch_bounds__(NTHREADS)
void gemm_bu_kernel(const unsigned short* __restrict__ Wbu, // (2P,H) bf16 panel
                    const float* __restrict__ u,            // (B,H,L)
                    float* __restrict__ xre, float* __restrict__ xim)
{
    __shared__ unsigned short sA[2][BM * TP];
    __shared__ unsigned short sB[BN * TP];

    const int tid  = threadIdx.x;
    const int lane = tid & 31;
    const int wave = tid >> 5;
    const int wm   = wave >> 2;          // 0..1
    const int wn   = wave & 3;           // 0..3
    const int l16  = lane & 15;
    const int hi   = (lane >> 4) & 1;

    const int b    = blockIdx.z;
    const int m0   = blockIdx.y * BM;    // 0..1023
    const int n0   = blockIdx.x * BN;
    const int comp = (m0 >= Pc) ? 1 : 0;
    const int p0   = m0 - comp * Pc;

    const v8f vzero = {0.f,0.f,0.f,0.f,0.f,0.f,0.f,0.f};
    v8f acc[4][2];
    #pragma unroll
    for (int i = 0; i < 4; ++i)
        #pragma unroll
        for (int j = 0; j < 2; ++j) acc[i][j] = vzero;

    const size_t uBase = ((size_t)b * Hc) * Lc + (size_t)n0;

#ifdef USE_TDM
    if (wave == 0)
        tdm_load_a_tile(Wbu, (unsigned)(uintptr_t)&sA[0][0], m0, 0, Hc, 2 * Pc);
#endif
    int cur = 0;

    for (int kt = 0; kt < Hc; kt += BKT) {
        __syncthreads();   // previous compute finished before sB is overwritten
        // Stage B tile transposed: sB[n][k] = bf16(u[b][kt+k][n0+n])
        #pragma unroll
        for (int it = 0; it < (BKT * BN) / NTHREADS; ++it) {
            int idx = tid + it * NTHREADS;
            int k = idx >> 7;            // /BN
            int n = idx & (BN - 1);
            float v = u[uBase + (size_t)(kt + k) * Lc + (size_t)n];
            sB[n * TP + k] = f2bf(v);
        }
        if (kt + BKT < Hc)  // prefetch next K-tile of u -> global_prefetch_b8
            __builtin_prefetch(&u[uBase + (size_t)(kt + BKT) * Lc + (size_t)(tid & 127)], 0, 1);
#ifdef USE_TDM
        if (wave == 0) {
            if (kt + BKT < Hc) {
                tdm_load_a_tile(Wbu, (unsigned)(uintptr_t)&sA[cur ^ 1][0],
                                m0, kt + BKT, Hc, 2 * Pc);
                __builtin_amdgcn_s_wait_tensorcnt(1);   // current tile landed
            } else {
                __builtin_amdgcn_s_wait_tensorcnt(0);
            }
        }
#else
        #pragma unroll
        for (int it = 0; it < (BM * BKT) / NTHREADS; ++it) {
            int idx = tid + it * NTHREADS;
            int m = idx >> 6;
            int k = idx & (BKT - 1);
            sA[cur][m * TP + k] = Wbu[(size_t)(m0 + m) * Hc + (size_t)(kt + k)];
        }
#endif
        __syncthreads();

        const unsigned short* sAc = &sA[cur][0];
        #pragma unroll
        for (int ks = 0; ks < BKT; ks += 32) {
            Frag fa[4], fb[2];
            #pragma unroll
            for (int mi = 0; mi < 4; ++mi) {
                int row = wm * 64 + mi * 16 + l16;
                const unsigned short* pa = &sAc[row * TP + ks + hi * 16];
                fa[mi].q[0] = *(const v4u*)(pa);        // K 0..7 (or 16..23)
                fa[mi].q[1] = *(const v4u*)(pa + 8);    // K 8..15 (or 24..31)
            }
            #pragma unroll
            for (int ni = 0; ni < 2; ++ni) {
                int col = wn * 32 + ni * 16 + l16;
                const unsigned short* pb = &sB[col * TP + ks + hi * 16];
                fb[ni].q[0] = *(const v4u*)(pb);        // 16 contiguous K
                fb[ni].q[1] = *(const v4u*)(pb + 8);
            }
            #pragma unroll
            for (int mi = 0; mi < 4; ++mi)
                #pragma unroll
                for (int ni = 0; ni < 2; ++ni)
                    acc[mi][ni] = __builtin_amdgcn_wmma_f32_16x16x32_bf16(
                        false, fa[mi].v, false, fb[ni].v,
                        (short)0, acc[mi][ni], false, false);
        }
        cur ^= 1;
    }

    float* plane = comp ? xim : xre;
    const size_t outBase = ((size_t)(b * Pc + p0)) * Lc;
    #pragma unroll
    for (int mi = 0; mi < 4; ++mi) {
        #pragma unroll
        for (int ni = 0; ni < 2; ++ni) {
            int gn = n0 + wn * 32 + ni * 16 + l16;
            #pragma unroll
            for (int r = 0; r < 8; ++r) {
                int gm = wm * 64 + mi * 16 + r + hi * 8;
                plane[outBase + (size_t)gm * Lc + (size_t)gn] = acc[mi][ni][r];
            }
        }
    }
}

// ---------------------------------------------------------------------------
// Kernel 2: chunked complex scan, in place on xre/xim.
// One wave per (b,p) sequence; 64 elements per lane; two passes + wave scan.
// ---------------------------------------------------------------------------
__global__ __launch_bounds__(NTHREADS)
void scan_kernel(const float* __restrict__ lamRI,
                 float* __restrict__ xre, float* __restrict__ xim)
{
    const int gwave = (int)((blockIdx.x * (unsigned)blockDim.x + threadIdx.x) >> 5);
    const int lane  = threadIdx.x & 31;
    const int b = gwave / Pc;
    const int p = gwave % Pc;

    const float lr = lamRI[2 * p + 0];
    const float li = lamRI[2 * p + 1];
    const size_t base = ((size_t)(b * Pc + p)) * Lc + (size_t)lane * 64;

    // Pass 1: local inclusive scan (zero init), store y_t in place.
    float xr = 0.f, xi = 0.f;
    for (int i = 0; i < 64; ++i) {
        float br = xre[base + i], bi = xim[base + i];
        float nr = lr * xr - li * xi + br;
        float ni = lr * xi + li * xr + bi;
        xr = nr; xi = ni;
        xre[base + i] = xr; xim[base + i] = xi;
    }
    // Chunk aggregate A = lam^64 via 6 complex squarings.
    float ar = lr, ai = li;
    #pragma unroll
    for (int s = 0; s < 6; ++s) {
        float nr = ar * ar - ai * ai;
        float ni = 2.f * ar * ai;
        ar = nr; ai = ni;
    }
    // Inclusive wave scan over (A, Bu) with the SSM binary op.
    float sAr = ar, sAi = ai, sBr = xr, sBi = xi;
    #pragma unroll
    for (int off = 1; off < 32; off <<= 1) {
        float pAr = __shfl_up(sAr, off, 32), pAi = __shfl_up(sAi, off, 32);
        float pBr = __shfl_up(sBr, off, 32), pBi = __shfl_up(sBi, off, 32);
        if (lane >= off) {
            float nBr = sAr * pBr - sAi * pBi + sBr;
            float nBi = sAr * pBi + sAi * pBr + sBi;
            float nAr = sAr * pAr - sAi * pAi;
            float nAi = sAr * pAi + sAi * pAr;
            sBr = nBr; sBi = nBi; sAr = nAr; sAi = nAi;
        }
    }
    float Xr = __shfl_up(sBr, 1, 32), Xi = __shfl_up(sBi, 1, 32);
    if (lane == 0) { Xr = 0.f; Xi = 0.f; }

    // Pass 2: x_t = y_t + lam^(t+1) * X
    float cr = lr, ci = li;
    for (int i = 0; i < 64; ++i) {
        float yr = xre[base + i], yi = xim[base + i];
        xre[base + i] = yr + cr * Xr - ci * Xi;
        xim[base + i] = yi + cr * Xi + ci * Xr;
        float nr = cr * lr - ci * li;
        float ni = cr * li + ci * lr;
        cr = nr; ci = ni;
    }
}

// ---------------------------------------------------------------------------
// Kernel 3: out = gelu( Cr@xre - Ci@xim + diag(D) .* u )
// K axis is 1024 over W_out = [Cr | -Ci]; B operand switches xre/xim planes.
// ---------------------------------------------------------------------------
__global__ __launch_bounds__(NTHREADS)
void gemm_out_kernel(const unsigned short* __restrict__ Wout, // (H,2P) bf16 panel
                     const float* __restrict__ xre, const float* __restrict__ xim,
                     const float* __restrict__ u,             // (B,H,L)
                     const float* __restrict__ Dm,            // (H,H)
                     float* __restrict__ out)                 // (B,H,L)
{
    __shared__ unsigned short sA[2][BM * TP];
    __shared__ unsigned short sB[BN * TP];

    const int tid  = threadIdx.x;
    const int lane = tid & 31;
    const int wave = tid >> 5;
    const int wm   = wave >> 2;
    const int wn   = wave & 3;
    const int l16  = lane & 15;
    const int hi   = (lane >> 4) & 1;

    const int b  = blockIdx.z;
    const int m0 = blockIdx.y * BM;      // 0..511 (H rows)
    const int n0 = blockIdx.x * BN;

    const v8f vzero = {0.f,0.f,0.f,0.f,0.f,0.f,0.f,0.f};
    v8f acc[4][2];
    #pragma unroll
    for (int i = 0; i < 4; ++i)
        #pragma unroll
        for (int j = 0; j < 2; ++j) acc[i][j] = vzero;

#ifdef USE_TDM
    if (wave == 0)
        tdm_load_a_tile(Wout, (unsigned)(uintptr_t)&sA[0][0], m0, 0, 2 * Pc, Hc);
#endif
    int cur = 0;

    for (int kt = 0; kt < 2 * Pc; kt += BKT) {
        const int comp = (kt >= Pc) ? 1 : 0;    // whole K-tile one component
        const int kp   = kt - comp * Pc;
        const float* plane = comp ? xim : xre;
        const size_t pBase = ((size_t)(b * Pc + kp)) * Lc + (size_t)n0;

        __syncthreads();
        #pragma unroll
        for (int it = 0; it < (BKT * BN) / NTHREADS; ++it) {
            int idx = tid + it * NTHREADS;
            int k = idx >> 7;
            int n = idx & (BN - 1);
            sB[n * TP + k] = f2bf(plane[pBase + (size_t)k * Lc + (size_t)n]);
        }
        if (kt + BKT < 2 * Pc)
            __builtin_prefetch(&plane[pBase + (size_t)BKT * Lc + (size_t)(tid & 127)], 0, 1);
#ifdef USE_TDM
        if (wave == 0) {
            if (kt + BKT < 2 * Pc) {
                tdm_load_a_tile(Wout, (unsigned)(uintptr_t)&sA[cur ^ 1][0],
                                m0, kt + BKT, 2 * Pc, Hc);
                __builtin_amdgcn_s_wait_tensorcnt(1);
            } else {
                __builtin_amdgcn_s_wait_tensorcnt(0);
            }
        }
#else
        #pragma unroll
        for (int it = 0; it < (BM * BKT) / NTHREADS; ++it) {
            int idx = tid + it * NTHREADS;
            int m = idx >> 6;
            int k = idx & (BKT - 1);
            sA[cur][m * TP + k] = Wout[(size_t)(m0 + m) * (2 * Pc) + (size_t)(kt + k)];
        }
#endif
        __syncthreads();

        const unsigned short* sAc = &sA[cur][0];
        #pragma unroll
        for (int ks = 0; ks < BKT; ks += 32) {
            Frag fa[4], fb[2];
            #pragma unroll
            for (int mi = 0; mi < 4; ++mi) {
                int row = wm * 64 + mi * 16 + l16;
                const unsigned short* pa = &sAc[row * TP + ks + hi * 16];
                fa[mi].q[0] = *(const v4u*)(pa);
                fa[mi].q[1] = *(const v4u*)(pa + 8);
            }
            #pragma unroll
            for (int ni = 0; ni < 2; ++ni) {
                int col = wn * 32 + ni * 16 + l16;
                const unsigned short* pb = &sB[col * TP + ks + hi * 16];
                fb[ni].q[0] = *(const v4u*)(pb);
                fb[ni].q[1] = *(const v4u*)(pb + 8);
            }
            #pragma unroll
            for (int mi = 0; mi < 4; ++mi)
                #pragma unroll
                for (int ni = 0; ni < 2; ++ni)
                    acc[mi][ni] = __builtin_amdgcn_wmma_f32_16x16x32_bf16(
                        false, fa[mi].v, false, fb[ni].v,
                        (short)0, acc[mi][ni], false, false);
        }
        cur ^= 1;
    }

    // Fused epilogue: + diag(D) .* u, exact GELU
    #pragma unroll
    for (int mi = 0; mi < 4; ++mi) {
        #pragma unroll
        for (int ni = 0; ni < 2; ++ni) {
            int gn = n0 + wn * 32 + ni * 16 + l16;
            #pragma unroll
            for (int r = 0; r < 8; ++r) {
                int gm = m0 + wm * 64 + mi * 16 + r + hi * 8;
                float dd = Dm[(size_t)gm * Hc + gm];
                size_t off = ((size_t)b * Hc + (size_t)gm) * Lc + (size_t)gn;
                float x = acc[mi][ni][r] + dd * u[off];
                out[off] = gelu_exact(x);
            }
        }
    }
}

// ---------------------------------------------------------------------------
extern "C" void kernel_launch(void* const* d_in, const int* in_sizes, int n_in,
                              void* d_out, int out_size, void* d_ws, size_t ws_size,
                              hipStream_t stream) {
    const float* u    = (const float*)d_in[0];   // (B,H,L)
    const float* lam  = (const float*)d_in[1];   // (P,2)
    const float* Bbar = (const float*)d_in[2];   // (P,H,2)
    const float* Cri  = (const float*)d_in[3];   // (H,P,2)
    const float* Dm   = (const float*)d_in[4];   // (H,H)
    float* out = (float*)d_out;

    const size_t planeElems = (size_t)Bc * Pc * Lc;          // 8.39M
    float* xre = (float*)d_ws;
    float* xim = xre + planeElems;
    unsigned short* Wbu  = (unsigned short*)(xim + planeElems);          // 2P*H halves
    unsigned short* Wout = Wbu + (size_t)(2 * Pc) * Hc;                  // H*2P halves

    // 0) pack weights to bf16 panels (once per call; tiny)
    pack_wbu_kernel<<<(2 * Pc * Hc) / NTHREADS, NTHREADS, 0, stream>>>(Bbar, Wbu);
    pack_wout_kernel<<<(Hc * 2 * Pc) / NTHREADS, NTHREADS, 0, stream>>>(Cri, Wout);

    // 1) Bu = B_bar @ u (re + im planes), bf16 WMMA GEMM, TDM-staged A tiles
    gemm_bu_kernel<<<dim3(Lc / BN, (2 * Pc) / BM, Bc), NTHREADS, 0, stream>>>(
        Wbu, u, xre, xim);

    // 2) complex associative scan over L, in place
    {
        int totalWaves = Bc * Pc;                          // 4096
        int blocks = (totalWaves * 32) / NTHREADS;         // 512
        scan_kernel<<<blocks, NTHREADS, 0, stream>>>(lam, xre, xim);
    }

    // 3) out = gelu(Cr@xre - Ci@xim + diag(D).*u), bf16 WMMA GEMM, fused epilogue
    gemm_out_kernel<<<dim3(Lc / BN, Hc / BM, Bc), NTHREADS, 0, stream>>>(
        Wout, xre, xim, u, Dm, out);
}
